// InverseAutoregressiveFlow_9655086481759
// MI455X (gfx1250) — matrix operand order repaired and checked by reading, not compiled, not run on hardware
//
#include <hip/hip_runtime.h>
#include <hip/hip_bf16.h>

// ---------------- problem constants (from reference setup_inputs) ----------
#define B_SZ   128
#define C_IN   16
#define C_OUT  8
#define SEQ_L  8192
#define T_FLOW 2
#define HID    32
#define KBINS  8
#define PPAR   23          // 3K-1
#define NPAR   184         // PPAR * C_OUT
#define TAILB  3.0f
#define MIN_W  1e-3f
#define MIN_H  1e-3f
#define MIN_D  1e-3f

// ---------------- tiling ---------------------------------------------------
#define TILE   128          // output positions per block
#define EXTENT 136          // loaded positions (TILE + 2*4 halo)
#define PADN   144          // padded compute width (9 x 16 WMMA N-tiles)
#define LDSW   146          // padded + 1 guard col each side
#define NWAVE  9
#define NTHR   (32*NWAVE)   // 288 threads
#define NTILES (SEQ_L / TILE)   // 64

typedef __attribute__((ext_vector_type(16))) __bf16 v16bf;
typedef __attribute__((ext_vector_type(8)))  __bf16 v8bf;
typedef __attribute__((ext_vector_type(8)))  float  v8f;

// ===========================================================================
// Fused 4-layer conv1d conditioner (bf16 WMMA, f32 accumulate).
// Activations stored position-major in LDS: hbuf[col][ch] (row = 64 B), so
// every WMMA B fragment is one contiguous 32-byte LDS load, D stores are one
// packed ds_store_b128 per M-tile. Weights staged [tap][co][ci] with K contiguous
// (same k = 16h+s order as B) so A fragments are also single contiguous loads.
// ===========================================================================
__global__ __launch_bounds__(NTHR)
void cond_kernel(const float* __restrict__ cin,
                 const float* __restrict__ xin,
                 const float* __restrict__ w0g,
                 const float* __restrict__ b0g,
                 const float* __restrict__ whg,
                 const float* __restrict__ bhg,
                 int t,
                 float* __restrict__ partial_h)
{
    const int tileIdx = blockIdx.x;          // 0..63
    const int b       = blockIdx.y;          // 0..127
    const int l0      = tileIdx * TILE;
    const int lane    = threadIdx.x;         // 0..31
    const int wy      = threadIdx.y;         // 0..8 (wave id)
    const int tid     = wy * 32 + lane;

    // position-major activations: [buffer][column q+1][channel]
    __shared__ __align__(16) __bf16 hbuf[2][LDSW][HID];
    // weights, K contiguous (slot = K): [tap d][co][slot]
    __shared__ __align__(16) __bf16 wlds[3][HID][HID];
    __shared__ float  blds[HID];

    // ---- zero both activation buffers (guards + pad columns) -------------
    {
        __bf16* p = &hbuf[0][0][0];
        for (int i = tid; i < 2 * LDSW * HID; i += NTHR) p[i] = (__bf16)0.0f;
    }
    __syncthreads();

    // ---- load layer-0 input: ch 0..15 = c, 16..23 = x, 24..31 = 0 --------
    for (int i = tid; i < HID * EXTENT; i += NTHR) {
        int chn = i / EXTENT;
        int q   = i - chn * EXTENT;          // 0..135
        int l   = l0 - 4 + q;
        float v = 0.0f;
        if (l >= 0 && l < SEQ_L) {
            if (chn < C_IN)            v = cin[(b * C_IN + chn) * SEQ_L + l];
            else if (chn < C_IN+C_OUT) v = xin[(b * C_OUT + (chn - C_IN)) * SEQ_L + l];
        }
        hbuf[0][q + 1][chn] = (__bf16)v;
    }
    __syncthreads();

    const bool edgeTile = (tileIdx == 0) || (tileIdx == NTILES - 1);
    const int  nn   = lane & 15;
    const int  half = lane >> 4;

    int cur = 0;
    for (int li = 0; li < 4; ++li) {
        const int nxt = cur ^ 1;

        // ---- stage weights/bias for this layer (f32 -> bf16, K contiguous) -
        const float* wsrc;
        const float* bsrc;
        int cin_l;
        if (li == 0) {
            wsrc  = w0g + (size_t)t * HID * (C_IN + C_OUT) * 3;
            bsrc  = b0g + (size_t)t * HID;
            cin_l = C_IN + C_OUT;            // 24
        } else {
            wsrc  = whg + (size_t)((t * 3) + (li - 1)) * HID * HID * 3;
            bsrc  = bhg + (size_t)((t * 3) + (li - 1)) * HID;
            cin_l = HID;                     // 32
        }
        for (int i = tid; i < 3 * HID * HID; i += NTHR) {
            int kk  = i / (HID * HID);
            int rem = i - kk * (HID * HID);
            int co  = rem / HID;
            int ci  = rem - co * HID;        // K index 0..31
            // A-fragment slot: K stored contiguously (slot = K)
            int s = ci;
            float v = (ci < cin_l) ? wsrc[((size_t)co * cin_l + ci) * 3 + kk] : 0.0f;
            wlds[kk][co][s] = (__bf16)v;
        }
        if (tid < HID) blds[tid] = bsrc[tid];
        __syncthreads();

        // ---- GEMM: out[32 x 144] = W[32 x 96] * im2col(in)[96 x 144] ----
        const int q0 = wy * 16;              // this wave's N-tile base

        // B fragments: lane (half,n) holds K = half*16+i at column n ->
        // contiguous channels at hbuf[col][half*16 ..]
        v16bf bfr[3];
        #pragma unroll
        for (int kk = 0; kk < 3; ++kk) {     // kk == conv tap d
            const int col = q0 + nn + kk;    // lds col of q0+nn + (d-1), +1 guard
            bfr[kk] = *(const v16bf*)&hbuf[cur][col][half * 16];
        }

        #pragma unroll
        for (int mt = 0; mt < 2; ++mt) {
            v8f acc = {};
            #pragma unroll
            for (int kk = 0; kk < 3; ++kk) {
                // A fragment: contiguous K = half*16 + i, same order as B
                const v16bf afr =
                    *(const v16bf*)&wlds[kk][mt * 16 + nn][half * 16];
                acc = __builtin_amdgcn_wmma_f32_16x16x32_bf16(
                    false, afr, false, bfr[kk], (short)0, acc, false, false);
            }
            // D layout: VGPR r, lanes 0-15 -> M=r, lanes 16-31 -> M=8+r.
            // Per lane: 8 consecutive channels starting at mt*16 + 8*half.
            const int colq = q0 + nn;
            const int co0  = mt * 16 + 8 * half;
            v8bf o;
            #pragma unroll
            for (int r = 0; r < 8; ++r) {
                float v = acc[r] + blds[co0 + r];
                v = v > 0.0f ? v : 0.0f;     // ReLU
                o[r] = (__bf16)v;
            }
            *(v8bf*)&hbuf[nxt][colq + 1][co0] = o;
        }
        __syncthreads();

        // ---- re-zero columns that fall outside the real sequence --------
        if (edgeTile) {
            for (int i = tid; i < LDSW * HID; i += NTHR) {
                int qq  = i / HID;
                int chn = i - qq * HID;
                int l   = l0 - 5 + qq;       // q = qq-1
                if (l < 0 || l >= SEQ_L) hbuf[nxt][qq][chn] = (__bf16)0.0f;
            }
        }
        __syncthreads();
        cur = nxt;
    }

    // ---- column-sum of final hidden layer over the 128 valid positions ---
    if (tid < HID) {
        float s = 0.0f;
        #pragma unroll 4
        for (int q = 4; q < 4 + TILE; ++q)
            s += (float)hbuf[cur][q + 1][tid];
        partial_h[((size_t)b * NTILES + tileIdx) * HID + tid] = s;
    }
}

// ===========================================================================
// Per-batch: finish the mean, apply the 1x1 out-conv (184x32 matvec), and
// build the RQS knot tables (cw[9], ch[9], derivs[9]) per channel.
// tables layout: [B][C_OUT][28]  (0..8 cw, 9..17 ch, 18..26 dv)
// ===========================================================================
__global__ __launch_bounds__(192)
void params_kernel(const float* __restrict__ partial_h,
                   const float* __restrict__ w_out,
                   const float* __restrict__ b_out,
                   int t,
                   float* __restrict__ tables)
{
    const int b   = blockIdx.x;
    const int tid = threadIdx.x;

    __shared__ float hm[HID];
    __shared__ float pbuf[NPAR];

    if (tid < HID) {
        float s = 0.0f;
        for (int tl = 0; tl < NTILES; ++tl)
            s += partial_h[((size_t)b * NTILES + tl) * HID + tid];
        hm[tid] = s / (float)SEQ_L;
    }
    __syncthreads();

    if (tid < NPAR) {
        const float* wr = w_out + ((size_t)t * NPAR + tid) * HID;
        float s = b_out[(size_t)t * NPAR + tid];
        #pragma unroll
        for (int i = 0; i < HID; ++i) s += wr[i] * hm[i];
        pbuf[tid] = s;
    }
    __syncthreads();

    if (tid < C_OUT) {
        const float* pw = &pbuf[tid * PPAR];
        float* tb = tables + ((size_t)b * C_OUT + tid) * 28;

        // widths -> knot x positions
        {
            float mx = pw[0];
            #pragma unroll
            for (int i = 1; i < KBINS; ++i) mx = fmaxf(mx, pw[i]);
            float e[KBINS], ssum = 0.0f;
            #pragma unroll
            for (int i = 0; i < KBINS; ++i) { e[i] = __expf(pw[i] - mx); ssum += e[i]; }
            float cum = 0.0f;
            tb[0] = -TAILB;
            #pragma unroll
            for (int i = 0; i < KBINS; ++i) {
                float w = MIN_W + (1.0f - MIN_W * KBINS) * e[i] / ssum;
                cum += w;
                tb[i + 1] = 2.0f * TAILB * cum - TAILB;
            }
        }
        // heights -> knot y positions
        {
            float mx = pw[KBINS];
            #pragma unroll
            for (int i = 1; i < KBINS; ++i) mx = fmaxf(mx, pw[KBINS + i]);
            float e[KBINS], ssum = 0.0f;
            #pragma unroll
            for (int i = 0; i < KBINS; ++i) { e[i] = __expf(pw[KBINS + i] - mx); ssum += e[i]; }
            float cum = 0.0f;
            tb[9] = -TAILB;
            #pragma unroll
            for (int i = 0; i < KBINS; ++i) {
                float h = MIN_H + (1.0f - MIN_H * KBINS) * e[i] / ssum;
                cum += h;
                tb[9 + i + 1] = 2.0f * TAILB * cum - TAILB;
            }
        }
        // derivatives (boundary = 1, interior = MIN_D + softplus)
        tb[18] = 1.0f;
        #pragma unroll
        for (int i = 0; i < KBINS - 1; ++i) {
            float d  = pw[2 * KBINS + i];
            float sp = (d > 20.0f) ? d : log1pf(__expf(d));
            tb[19 + i] = MIN_D + sp;
        }
        tb[26] = 1.0f;
    }
}

// ===========================================================================
// Elementwise RQS inverse + per-block log-det partial sum.
// grid = (L/256, C_OUT, B); one (b,channel) spline table per block.
// ===========================================================================
__global__ __launch_bounds__(256)
void rqs_kernel(float* __restrict__ x,
                const float* __restrict__ tables,
                float* __restrict__ partial_ld)
{
    const int b   = blockIdx.z;
    const int chn = blockIdx.y;
    const int tid = threadIdx.x;

    __shared__ float tsm[27];
    __shared__ float red[256];

    if (tid < 27) tsm[tid] = tables[((size_t)b * C_OUT + chn) * 28 + tid];
    __syncthreads();
    const float* cw = tsm;
    const float* ch = tsm + 9;
    const float* dv = tsm + 18;

    const size_t base = ((size_t)b * C_OUT + chn) * SEQ_L;
    const int    l    = blockIdx.x * 256 + tid;

    float y = x[base + l];
    const bool inside = fabsf(y) <= TAILB;
    float yc = fminf(fmaxf(y, -TAILB), TAILB);

    // bin search over K+1 knot y-positions (K = 8, broadcast compare)
    int idx = -1;
    #pragma unroll
    for (int i = 0; i < KBINS + 1; ++i) idx += (yc >= ch[i]) ? 1 : 0;
    idx = min(max(idx, 0), KBINS - 1);

    const float in_cw = cw[idx];
    const float in_w  = cw[idx + 1] - cw[idx];
    const float in_ch = ch[idx];
    const float in_h  = ch[idx + 1] - ch[idx];
    const float dk    = dv[idx];
    const float dk1   = dv[idx + 1];

    const float s   = in_h / in_w;
    const float dy  = yc - in_ch;
    const float t1  = dk + dk1 - 2.0f * s;
    const float aa  = in_h * (s - dk) + dy * t1;
    const float bb  = in_h * dk - dy * t1;
    const float cq  = -s * dy;
    const float dis = fmaxf(bb * bb - 4.0f * aa * cq, 0.0f);
    const float root = (2.0f * cq) / (-bb - sqrtf(dis));
    const float outv = root * in_w + in_cw;
    const float om   = 1.0f - root;
    const float den  = s + t1 * root * om;
    const float dnum = s * s * (dk1 * root * root + 2.0f * s * root * om + dk * om * om);
    const float ld   = -(__logf(dnum) - 2.0f * __logf(den));

    x[base + l] = inside ? outv : y;
    red[tid]    = inside ? ld : 0.0f;
    __syncthreads();

    for (int stp = 128; stp > 0; stp >>= 1) {
        if (tid < stp) red[tid] += red[tid + stp];
        __syncthreads();
    }
    if (tid == 0)
        partial_ld[((size_t)b * C_OUT + chn) * (SEQ_L / 256) + blockIdx.x] = red[0];
}

// Reduce 8*32 = 256 partials per batch, accumulate into log_det output.
__global__ __launch_bounds__(256)
void ld_reduce_kernel(const float* __restrict__ partial_ld,
                      float* __restrict__ ldout)
{
    const int b   = blockIdx.x;
    const int tid = threadIdx.x;
    __shared__ float red[256];
    red[tid] = partial_ld[(size_t)b * 256 + tid];
    __syncthreads();
    for (int stp = 128; stp > 0; stp >>= 1) {
        if (tid < stp) red[tid] += red[tid + stp];
        __syncthreads();
    }
    if (tid == 0) ldout[b] += red[0] / (float)SEQ_L;
}

__global__ void init_x_kernel(const float* __restrict__ noise,
                              float* __restrict__ xout, int n)
{
    int i = blockIdx.x * blockDim.x + threadIdx.x;
    if (i < n) xout[i] = noise[i];
}

__global__ void zero_ld_kernel(float* __restrict__ ldout)
{
    ldout[threadIdx.x] = 0.0f;
}

// ===========================================================================
extern "C" void kernel_launch(void* const* d_in, const int* in_sizes, int n_in,
                              void* d_out, int out_size, void* d_ws, size_t ws_size,
                              hipStream_t stream)
{
    (void)in_sizes; (void)n_in; (void)out_size; (void)ws_size;

    const float* cin   = (const float*)d_in[1];
    const float* noise = (const float*)d_in[2];
    const float* w0g   = (const float*)d_in[3];
    const float* b0g   = (const float*)d_in[4];
    const float* whg   = (const float*)d_in[5];
    const float* bhg   = (const float*)d_in[6];
    const float* w_out = (const float*)d_in[7];
    const float* b_out = (const float*)d_in[8];

    float* xout  = (float*)d_out;                         // [B, C_OUT, L]
    float* ldout = xout + (size_t)B_SZ * C_OUT * SEQ_L;   // [B, 1]

    float* ws         = (float*)d_ws;
    float* partial_h  = ws;                                      // B*64*32
    float* tables     = partial_h + (size_t)B_SZ * NTILES * HID; // B*8*28
    float* partial_ld = tables + (size_t)B_SZ * C_OUT * 28;      // B*8*32

    const int nx = B_SZ * C_OUT * SEQ_L;
    init_x_kernel<<<(nx + 255) / 256, 256, 0, stream>>>(noise, xout, nx);
    zero_ld_kernel<<<1, B_SZ, 0, stream>>>(ldout);

    for (int t = 0; t < T_FLOW; ++t) {
        cond_kernel<<<dim3(NTILES, B_SZ), dim3(32, NWAVE), 0, stream>>>(
            cin, xout, w0g, b0g, whg, bhg, t, partial_h);
        params_kernel<<<B_SZ, 192, 0, stream>>>(partial_h, w_out, b_out, t, tables);
        rqs_kernel<<<dim3(SEQ_L / 256, C_OUT, B_SZ), 256, 0, stream>>>(
            xout, tables, partial_ld);
        ld_reduce_kernel<<<B_SZ, 256, 0, stream>>>(partial_ld, ldout);
    }
}
